// S4ND_9431748182375
// MI455X (gfx1250) — matrix operand order repaired and checked by reading, not compiled, not run on hardware
//
#include <hip/hip_runtime.h>

// ---------------------------------------------------------------------------
// S4ND as separable bidirectional 2-D conv via Toeplitz GEMMs on WMMA bf16.
//   y[b,h] = T1_h * U[b,h] * T2_h^T + D[h]*U[b,h]
// T[i][j] = kf[i-j] (i>=j) else kb[j-i-1], kf/kb = S4D ZOH kernels (len 256).
// Pass 4 computes V = U*T2^T and stores V^T; pass 5 computes Y^T = V^T*T1^T so
// both the WMMA B-operands and the f32 output/skip-term accesses are fully
// contiguous b128 transactions.
// ---------------------------------------------------------------------------

typedef __bf16        v16bf __attribute__((ext_vector_type(16)));
typedef float         v8f   __attribute__((ext_vector_type(8)));
typedef float         f32x4 __attribute__((ext_vector_type(4)));
typedef unsigned int  u32x4 __attribute__((ext_vector_type(4)));

#define Hh 128
#define Nn 64
#define Ll 256
#define Bb 2
#define LL 65536              // 256*256

// workspace layout (bytes)
#define OFF_UBF 0u                                  // u in bf16: 2*128*64K*2 = 32 MiB
#define OFF_T1  33554432u                           // T1: 128*64K*2 = 16 MiB
#define OFF_T2  50331648u                           // T2: 16 MiB
#define OFF_VT  67108864u                           // V^T bf16: 32 MiB
#define OFF_KB  100663296u                          // k kernels f32: 4*128*256*4 = 512 KiB

union Frag {
    u32x4 q[2];
    v16bf v;
};

__device__ __forceinline__ unsigned short f2bf(float f) {
    unsigned int x = __float_as_uint(f);
    x += 0x7FFFu + ((x >> 16) & 1u);                // round-to-nearest-even
    return (unsigned short)(x >> 16);
}

__device__ __forceinline__ u32x4 pack8bf(v8f c) {
    u32x4 r;
    r.x = (unsigned)f2bf(c[0]) | ((unsigned)f2bf(c[1]) << 16);
    r.y = (unsigned)f2bf(c[2]) | ((unsigned)f2bf(c[3]) << 16);
    r.z = (unsigned)f2bf(c[4]) | ((unsigned)f2bf(c[5]) << 16);
    r.w = (unsigned)f2bf(c[6]) | ((unsigned)f2bf(c[7]) << 16);
    return r;
}

// A fragment: 16x32 bf16 tile, rows mbase..+15, cols kb..+31, row-major src.
// ISA layout: lane<16 holds K[0..7],K[16..23]; lane>=16 holds K[8..15],K[24..31].
__device__ __forceinline__ Frag load_a(const unsigned short* __restrict__ p,
                                       int mbase, int kb, int lane) {
    const int half = lane >> 4;
    const unsigned short* r = p + (size_t)(mbase + (lane & 15)) * 256 + kb + half * 8;
    Frag f;
    f.q[0] = *(const u32x4*)r;
    f.q[1] = *(const u32x4*)(r + 16);
    return f;
}

// B fragment: 32x16 with B[k][n] = src[n][kb + k] (src row-major over n).
// lane holds column n = nbase+(lane&15); lane<16 holds K[0..7],K[16..23]; lane>=16 holds K[8..15],K[24..31].
__device__ __forceinline__ Frag load_b(const unsigned short* __restrict__ p,
                                       int nbase, int kb, int lane) {
    const int half = lane >> 4;
    const unsigned short* r = p + (size_t)(nbase + (lane & 15)) * 256 + kb + half * 8;
    Frag f;
    f.q[0] = *(const u32x4*)r;
    f.q[1] = *(const u32x4*)(r + 16);
    return f;
}

// ---------------- pass 1: f32 -> bf16 convert of u ----------------
__global__ __launch_bounds__(256) void k_cvt(const float* __restrict__ u,
                                             unsigned short* __restrict__ ubf) {
    size_t i = (size_t)blockIdx.x * 256 + threadIdx.x;
    ubf[i] = f2bf(u[i]);
}

// ---------------- pass 2: S4D ZOH kernel generation ----------------
// grid: bit8 = axis, bit7 = channel (fwd/bwd), bits0-6 = h ; thread = l
__global__ __launch_bounds__(256) void k_ssm(
    const float* __restrict__ Ar0, const float* __restrict__ Ai0,
    const float* __restrict__ Cr0, const float* __restrict__ Ci0,
    const float* __restrict__ ldt0,
    const float* __restrict__ Ar1, const float* __restrict__ Ai1,
    const float* __restrict__ Cr1, const float* __restrict__ Ci1,
    const float* __restrict__ ldt1,
    float* __restrict__ kbuf) {
    const int h    = blockIdx.x & 127;
    const int c    = (blockIdx.x >> 7) & 1;
    const int axis = blockIdx.x >> 8;
    const float* Ar  = axis ? Ar1  : Ar0;
    const float* Ai  = axis ? Ai1  : Ai0;
    const float* Cr  = axis ? Cr1  : Cr0;
    const float* Ci  = axis ? Ci1  : Ci0;
    const float* ldt = axis ? ldt1 : ldt0;

    const int   l  = threadIdx.x;
    const float fl = (float)l;
    const float dt = __expf(ldt[h]);

    float acc = 0.f;
    for (int n = 0; n < Nn; ++n) {
        const float ar = Ar[h * Nn + n], ai = Ai[h * Nn + n];
        const float cr = Cr[((size_t)c * Hh + h) * Nn + n];
        const float ci = Ci[((size_t)c * Hh + h) * Nn + n];
        const float dar = dt * ar, dai = dt * ai;
        // g = (exp(dtA) - 1) / A
        float e = __expf(dar), s, co;
        __sincosf(dai, &s, &co);
        const float exr = e * co - 1.0f, exi = e * s;
        const float den = 1.0f / (ar * ar + ai * ai);
        const float gr = (exr * ar + exi * ai) * den;
        const float gi = (exi * ar - exr * ai) * den;
        // w = C * g
        const float wr = cr * gr - ci * gi;
        const float wi = cr * gi + ci * gr;
        // z = exp(dtA * l)
        float el = __expf(dar * fl), sl, cl;
        __sincosf(dai * fl, &sl, &cl);
        acc += 2.0f * (wr * el * cl - wi * el * sl);
    }
    kbuf[(((size_t)axis * 2 + c) * Hh + h) * Ll + l] = acc;
}

// ---------------- pass 3: build Toeplitz matrices (bf16) ----------------
__global__ __launch_bounds__(256) void k_toeplitz(const float* __restrict__ kbuf,
                                                  unsigned short* __restrict__ T1,
                                                  unsigned short* __restrict__ T2) {
    const unsigned idx = blockIdx.x * 256u + threadIdx.x;   // < 2*128*65536
    const int axis = idx >> 23;
    const unsigned r = idx & ((1u << 23) - 1);
    const int h = r >> 16;
    const int i = (r >> 8) & 255;
    const int j = r & 255;
    const float* kf = kbuf + (((size_t)axis * 2 + 0) * Hh + h) * Ll;
    const float* kb = kbuf + (((size_t)axis * 2 + 1) * Hh + h) * Ll;
    const float v = (i >= j) ? kf[i - j] : kb[j - i - 1];
    unsigned short* dst = axis ? T2 : T1;
    dst[(size_t)h * LL + (size_t)i * 256 + j] = f2bf(v);
}

// ---------------- pass 4: V = U * T2^T  (conv along L2), store V^T ----------
__global__ __launch_bounds__(256) void k_conv_l2(const unsigned short* __restrict__ ubf,
                                                 const unsigned short* __restrict__ T2,
                                                 unsigned short* __restrict__ VT) {
    const int bh     = blockIdx.x >> 1;
    const int rowblk = blockIdx.x & 1;
    const int h      = bh & (Hh - 1);
    const int wave = threadIdx.x >> 5;
    const int lane = threadIdx.x & 31;
    const int half = lane >> 4;

    const unsigned short* A  = ubf + (size_t)bh * LL;
    const unsigned short* Bm = T2  + (size_t)h  * LL;
    unsigned short*       Vo = VT  + (size_t)bh * LL;

    const int mbase = rowblk * 128 + (wave >> 2) * 64;   // m = x (L1 row)
    const int nbase = (wave & 3) * 64;                   // n = y (L2 col)

    v8f acc[4][4] = {};
    for (int kb = 0; kb < 256; kb += 32) {
        Frag fa[4], fb[4];
#pragma unroll
        for (int i = 0; i < 4; ++i) fa[i] = load_a(A, mbase + i * 16, kb, lane);
#pragma unroll
        for (int j = 0; j < 4; ++j) fb[j] = load_b(Bm, nbase + j * 16, kb, lane);
#pragma unroll
        for (int i = 0; i < 4; ++i)
#pragma unroll
            for (int j = 0; j < 4; ++j)
                acc[i][j] = __builtin_amdgcn_wmma_f32_16x16x32_bf16(
                    false, fa[i].v, false, fb[j].v, (short)0, acc[i][j], false, false);
    }
    // store transposed: C[m][n] -> VT[n][m]; 8 consecutive m per lane = one b128
#pragma unroll
    for (int i = 0; i < 4; ++i)
#pragma unroll
        for (int j = 0; j < 4; ++j) {
            const int nrow = nbase + j * 16 + (lane & 15);
            const int mcol = mbase + i * 16 + half * 8;
            *(u32x4*)(Vo + (size_t)nrow * 256 + mcol) = pack8bf(acc[i][j]);
        }
}

// ---------------- pass 5: Y^T = V^T * T1^T + D*u  (conv along L1) ------------
// C(M=y, N=i) = A(V^T: rows y, k=x) x B(B[k=x][n=i] = T1[i][x]).
// Per-lane contiguous-M axis of the C fragment == y, the contiguous axis of
// the (b,h,L1,L2) output -> vectorized f32 b128 loads of u and stores of y.
__global__ __launch_bounds__(256) void k_conv_l1(const unsigned short* __restrict__ T1,
                                                 const unsigned short* __restrict__ VT,
                                                 const float* __restrict__ u,
                                                 const float* __restrict__ D,
                                                 float* __restrict__ out) {
    const int bh     = blockIdx.x >> 1;
    const int rowblk = blockIdx.x & 1;
    const int h      = bh & (Hh - 1);
    const int wave = threadIdx.x >> 5;
    const int lane = threadIdx.x & 31;
    const int half = lane >> 4;

    const unsigned short* A  = VT + (size_t)bh * LL;   // V^T, rows y, contiguous x
    const unsigned short* Bm = T1 + (size_t)h  * LL;   // B[k=x][n=i] = T1[i][x]
    const float*          ub = u  + (size_t)bh * LL;
    float*                ob = out + (size_t)bh * LL;
    const float dh = D[h];

    const int mbase = rowblk * 128 + (wave >> 2) * 64;   // m = y
    const int nbase = (wave & 3) * 64;                   // n = i

    v8f acc[4][4] = {};
    for (int kb = 0; kb < 256; kb += 32) {
        Frag fa[4], fb[4];
#pragma unroll
        for (int t = 0; t < 4; ++t) fa[t] = load_a(A, mbase + t * 16, kb, lane);
#pragma unroll
        for (int j = 0; j < 4; ++j) fb[j] = load_b(Bm, nbase + j * 16, kb, lane);
#pragma unroll
        for (int t = 0; t < 4; ++t)
#pragma unroll
            for (int j = 0; j < 4; ++j)
                acc[t][j] = __builtin_amdgcn_wmma_f32_16x16x32_bf16(
                    false, fa[t].v, false, fb[j].v, (short)0, acc[t][j], false, false);
    }
    // epilogue: out[i][y0..y0+7] = acc + D[h]*u[i][y0..y0+7], two b128 each
#pragma unroll
    for (int t = 0; t < 4; ++t)
#pragma unroll
        for (int j = 0; j < 4; ++j) {
            const int irow = nbase + j * 16 + (lane & 15);
            const int y0   = mbase + t * 16 + half * 8;
            const size_t base = (size_t)irow * 256 + y0;
            const f32x4 lo = __builtin_shufflevector(acc[t][j], acc[t][j], 0, 1, 2, 3);
            const f32x4 hi = __builtin_shufflevector(acc[t][j], acc[t][j], 4, 5, 6, 7);
            const f32x4 u0 = *(const f32x4*)(ub + base);
            const f32x4 u1 = *(const f32x4*)(ub + base + 4);
            *(f32x4*)(ob + base)     = lo + dh * u0;
            *(f32x4*)(ob + base + 4) = hi + dh * u1;
        }
}

extern "C" void kernel_launch(void* const* d_in, const int* in_sizes, int n_in,
                              void* d_out, int out_size, void* d_ws, size_t ws_size,
                              hipStream_t stream) {
    (void)in_sizes; (void)n_in; (void)out_size; (void)ws_size;
    const float* u    = (const float*)d_in[0];
    const float* D    = (const float*)d_in[1];
    const float* Ar0  = (const float*)d_in[2];
    const float* Ai0  = (const float*)d_in[3];
    const float* Cr0  = (const float*)d_in[4];
    const float* Ci0  = (const float*)d_in[5];
    const float* ldt0 = (const float*)d_in[6];
    const float* Ar1  = (const float*)d_in[7];
    const float* Ai1  = (const float*)d_in[8];
    const float* Cr1  = (const float*)d_in[9];
    const float* Ci1  = (const float*)d_in[10];
    const float* ldt1 = (const float*)d_in[11];
    float* out = (float*)d_out;

    char* ws = (char*)d_ws;
    unsigned short* ubf  = (unsigned short*)(ws + OFF_UBF);
    unsigned short* T1   = (unsigned short*)(ws + OFF_T1);
    unsigned short* T2   = (unsigned short*)(ws + OFF_T2);
    unsigned short* VT   = (unsigned short*)(ws + OFF_VT);
    float*          kbuf = (float*)(ws + OFF_KB);

    k_cvt<<<Bb * Hh * LL / 256, 256, 0, stream>>>(u, ubf);
    k_ssm<<<2 * 2 * Hh, 256, 0, stream>>>(Ar0, Ai0, Cr0, Ci0, ldt0,
                                          Ar1, Ai1, Cr1, Ci1, ldt1, kbuf);
    k_toeplitz<<<2 * Hh * LL / 256, 256, 0, stream>>>(kbuf, T1, T2);
    k_conv_l2<<<Bb * Hh * 2, 256, 0, stream>>>(ubf, T2, VT);
    k_conv_l1<<<Bb * Hh * 2, 256, 0, stream>>>(T1, VT, u, D, out);
}
